// Block_56650618634972
// MI455X (gfx1250) — compile-verified, matrix operand order checked by reading, not run on hardware
//
#include <hip/hip_runtime.h>
#include <hip/hip_bf16.h>

// ---------------------------------------------------------------------------
// Strategy (MI455X / gfx1250, wave32):
//   The 8 butterfly stages + BNs are affine in the 256-channel vector =>
//   fold into a single 256x256 matrix M and bias (kernel 1, 1 block).
//   Then one fused memory-bound pass (kernel 2):
//     depthwise 3x3 conv + BN + ReLU  ->  LDS f16 tile (56x256, padded)
//     -> v_wmma_f32_16x16x32_f16 channel-GEMM (f32 accum) -> bias+ReLU -> out.
//   HBM traffic ~ read x (103MB) + write out (103MB) => ~9us floor @23.3TB/s.
//   M (128KB as f16) is L2/L0 resident; activations staged through LDS.
// ---------------------------------------------------------------------------

typedef __attribute__((ext_vector_type(16))) _Float16     v16h;
typedef __attribute__((ext_vector_type(8)))  float        v8f;
typedef __attribute__((ext_vector_type(4)))  unsigned int v4u;

union Frag { v16h v; v4u q[2]; };

struct BFParams {
    const float* bw[8];
    const float* bg[8];
    const float* bb[8];
    const float* bm[8];
    const float* bv[8];
};

// K index within a 16-half WMMA A/B fragment, per CDNA5 ISA 16-bit layout:
// lanes 0-15: halves 0..7 -> K 0..7,  halves 8..15 -> K 16..23
// lanes 16-31: halves 0..7 -> K 8..15, halves 8..15 -> K 24..31
__device__ __forceinline__ int frag_k(int lane, int j) {
    return ((j >> 3) << 4) + ((lane >> 4) << 3) + (j & 7);
}

// ---------------------------------------------------------------------------
// Kernel 1: build affine map (M, bias) from butterfly weights + BNs; pack M
// into f16 WMMA-B fragment order so the GEMM does per-lane contiguous loads.
// Single block of 256 threads; thread tid owns output-channel row tid.
// ---------------------------------------------------------------------------
__global__ __launch_bounds__(256) void build_affine(
    BFParams P,
    float* __restrict__ Ma, float* __restrict__ Mb,
    float* __restrict__ Ba, float* __restrict__ Bb,
    _Float16* __restrict__ Mh, float* __restrict__ biasF)
{
    const int tid = threadIdx.x;
    float* cur = Ma; float* nxt = Mb;
    float* bc  = Ba; float* bn_ = Bb;

    // identity init
    for (int c = 0; c < 256; ++c) cur[tid * 256 + c] = (c == tid) ? 1.f : 0.f;
    bc[tid] = 0.f;
    __syncthreads();

    for (int s = 0; s < 8; ++s) {
        const int Pd  = 128 >> s;          // output rows per group
        const int q   = tid / Pd;          // 2*grp + k  (BN channel index)
        const int p   = tid - q * Pd;
        const int grp = q >> 1, k = q & 1;
        const int i0  = grp * (Pd << 1) + (p << 1);
        const int i1  = i0 + 1;
        const float w0 = P.bw[s][grp * 4 + k * 2 + 0];
        const float w1 = P.bw[s][grp * 4 + k * 2 + 1];
        const float sc = P.bg[s][q] / sqrtf(P.bv[s][q] + 1e-5f);
        const float sh = P.bb[s][q] - P.bm[s][q] * sc;

        const float* r0 = cur + i0 * 256;
        const float* r1 = cur + i1 * 256;
        float*       rn = nxt + tid * 256;
        for (int c = 0; c < 256; ++c)
            rn[c] = sc * (w0 * r0[c] + w1 * r1[c]);
        bn_[tid] = sc * (w0 * bc[i0] + w1 * bc[i1]) + sh;
        __syncthreads();
        float* t;
        t = cur; cur = nxt; nxt = t;
        t = bc;  bc  = bn_; bn_ = t;
    }

    biasF[tid] = bc[tid];

    // Pack M -> f16 B-fragments: Mh[(((ks*16+ntile)*32)+lane)*16 + j]
    //   column co = ntile*16 + (lane&15), row ci = ks*32 + frag_k(lane, j)
    for (int e = tid; e < 65536; e += 256) {
        const int j     = e & 15;
        const int lane  = (e >> 4) & 31;
        const int ntile = (e >> 9) & 15;
        const int ks    = e >> 13;
        const int co    = ntile * 16 + (lane & 15);
        const int ci    = (ks << 5) + frag_k(lane, j);
        Mh[e] = (_Float16)cur[co * 256 + ci];
    }
}

// ---------------------------------------------------------------------------
// Kernel 2: per (n, h) row: conv+BN+ReLU into LDS (f16), then WMMA GEMM.
// Grid: (56, 32); block: 256 threads = 8 waves.
// ---------------------------------------------------------------------------
#define VROW 264  // 256 + 8 halves pad: kills LDS bank conflicts, keeps 16B align

__global__ __launch_bounds__(256) void fused_butterfly(
    const float* __restrict__ x,  const float* __restrict__ dw,
    const float* __restrict__ g1, const float* __restrict__ b1,
    const float* __restrict__ m1, const float* __restrict__ v1,
    const _Float16* __restrict__ Mh, const float* __restrict__ biasF,
    float* __restrict__ out)
{
    __shared__ _Float16 v_lds[64 * VROW];   // 33,792 B

    const int h   = blockIdx.x;
    const int n   = blockIdx.y;
    const int tid = threadIdx.x;

    // ---------------- Phase A: depthwise conv 3x3 + BN + ReLU -> LDS -------
    const int wq = tid & 63;   // spatial position (56 valid + 8 zero-pad)
    const int cq = tid >> 6;   // channel phase (4)
    for (int it = 0; it < 64; ++it) {
        const int ci = cq + (it << 2);
        float val = 0.f;
        if (wq < 56) {
            const float* xc = x + (((size_t)n * 256 + ci) * 3136);
            const float* wk = dw + ci * 9;
            float acc = 0.f;
#pragma unroll
            for (int dy = -1; dy <= 1; ++dy) {
                const int hh = h + dy;
                if (hh >= 0 && hh < 56) {
                    const float* xr = xc + hh * 56;
#pragma unroll
                    for (int dx = -1; dx <= 1; ++dx) {
                        const int ww = wq + dx;
                        if (ww >= 0 && ww < 56)
                            acc += wk[(dy + 1) * 3 + (dx + 1)] * xr[ww];
                    }
                }
            }
            const float inv = g1[ci] / sqrtf(v1[ci] + 1e-5f);
            const float bet = b1[ci] - m1[ci] * inv;
            val = acc * inv + bet;
            val = val > 0.f ? val : 0.f;
        }
        v_lds[wq * VROW + ci] = (_Float16)val;   // rows 56..63 zero-padded
    }
    __syncthreads();

    // ---------------- Phase B: 256-wide channel GEMM via WMMA --------------
    const int wave = tid >> 5;
    const int lane = tid & 31;
    const int lr   = lane & 15;
    const int hi8  = (lane >> 4) << 3;

    for (int tile = wave; tile < 64; tile += 8) {     // 4 m-tiles x 16 n-tiles
        const int m0    = (tile >> 4) << 4;           // spatial base
        const int ntile = tile & 15;
        const int n0    = ntile << 4;                 // out-channel base

        v8f acc = {};
#pragma unroll
        for (int ks = 0; ks < 8; ++ks) {              // K = 256 in steps of 32
            Frag fa, fb;
            // A: activation tile from LDS, documented 16-bit A layout
            const _Float16* ap = &v_lds[(m0 + lr) * VROW + (ks << 5) + hi8];
            fa.q[0] = *(const v4u*)(ap);              // K +0..7
            fa.q[1] = *(const v4u*)(ap + 16);         // K +16..23
            // B: pre-swizzled M fragments, per-lane contiguous 32B (L2 hot)
            const _Float16* bp = Mh + ((((ks << 4) + ntile) << 5) + lane) * 16;
            fb.q[0] = *(const v4u*)(bp);
            fb.q[1] = *(const v4u*)(bp + 8);

            acc = __builtin_amdgcn_wmma_f32_16x16x32_f16(
                false, fa.v, false, fb.v, (short)0, acc, false, false);
        }

        // bias + final ReLU, scatter to NCHW (rows of 56 per channel)
        const int   co   = n0 + lr;
        const float bvv  = biasF[co];
        float* obase = out + (((size_t)n * 256 + co) * 56 + h) * 56;
#pragma unroll
        for (int r = 0; r < 8; ++r) {
            const int w = m0 + hi8 + r;               // D: M = r (+8 hi half)
            if (w < 56) {
                float t = acc[r] + bvv;
                obase[w] = t > 0.f ? t : 0.f;
            }
        }
    }
}

// ---------------------------------------------------------------------------
extern "C" void kernel_launch(void* const* d_in, const int* in_sizes, int n_in,
                              void* d_out, int out_size, void* d_ws, size_t ws_size,
                              hipStream_t stream)
{
    const float* x  = (const float*)d_in[0];
    const float* dw = (const float*)d_in[1];
    const float* g1 = (const float*)d_in[2];
    const float* b1 = (const float*)d_in[3];
    const float* m1 = (const float*)d_in[4];
    const float* v1 = (const float*)d_in[5];

    BFParams bp;
    if (n_in >= 46) {
        // lists flattened as separate inputs: bw0..7, bg0..7, bb0..7, bm0..7, bv0..7
        for (int i = 0; i < 8; ++i) {
            bp.bw[i] = (const float*)d_in[6 + i];
            bp.bg[i] = (const float*)d_in[14 + i];
            bp.bb[i] = (const float*)d_in[22 + i];
            bp.bm[i] = (const float*)d_in[30 + i];
            bp.bv[i] = (const float*)d_in[38 + i];
        }
    } else {
        // lists concatenated: d_in[6]=bw(1020), d_in[7..10]=bg/bb/bm/bv(510 each)
        const float* bwc = (const float*)d_in[6];
        const float* bgc = (const float*)d_in[7];
        const float* bbc = (const float*)d_in[8];
        const float* bmc = (const float*)d_in[9];
        const float* bvc = (const float*)d_in[10];
        int ow = 0, ob = 0;
        for (int i = 0; i < 8; ++i) {
            bp.bw[i] = bwc + ow;  ow += 4 << i;       // (2^i, 2, 2)
            bp.bg[i] = bgc + ob;
            bp.bb[i] = bbc + ob;
            bp.bm[i] = bmc + ob;
            bp.bv[i] = bvc + ob;  ob += 2 << i;       // (2^{i+1},)
        }
    }

    // workspace layout (floats): Ma[65536] Mb[65536] Ba[256] Bb[256]
    //                            Mh (65536 f16 = 32768 f32 slots) biasF[256]
    float*     ws    = (float*)d_ws;
    float*     Ma    = ws;
    float*     Mb    = ws + 65536;
    float*     Ba    = ws + 131072;
    float*     Bb    = ws + 131328;
    _Float16*  Mh    = (_Float16*)(ws + 131584);
    float*     biasF = ws + 131584 + 32768;

    build_affine<<<1, 256, 0, stream>>>(bp, Ma, Mb, Ba, Bb, Mh, biasF);

    dim3 grid(56, 32);
    fused_butterfly<<<grid, 256, 0, stream>>>(x, dw, g1, b1, m1, v1,
                                              Mh, biasF, (float*)d_out);
}
